// CustomMamba_61392262529318
// MI455X (gfx1250) — matrix operand order repaired and checked by reading, not compiled, not run on hardware
//
#include <hip/hip_runtime.h>
#include <hip/hip_bf16.h>
#include <math.h>

typedef __bf16 bf16_t;
typedef bf16_t v16bf __attribute__((ext_vector_type(16)));
typedef float  v8f   __attribute__((ext_vector_type(8)));

#define Bn    32
#define Fn    64
#define Nn    207
#define Tn    24
#define DIN   128
#define DST   16
#define DTR   4
#define DCONV 4

// ---- LDS arena (phase-aliased), total 63488 bytes ----
// All WMMA store targets are padded to 32 rows -> unconditional stores.
// All WMMA A-operand sources are 32-row zero-padded bf16 -> unconditional loads.
#define OFF_H     0        // 24*64*4   = 6144   residual h (persistent)
#define OFF_XBF   6144     // 32*128*2  = 8192   x from in_proj (bf16)
#define OFF_ZSIL  14336    // 32*128*2  = 8192   silu(z) (bf16, fused at store)
#define OFF_UF    22528    // 32*128*2  = 8192   u_f = silu(conv_f(x)) bf16, pad zeroed
#define OFF_UB    30720    // 32*128*2  = 8192   u_b bf16, pad zeroed
#define OFF_HNBF  38912    // 32*64*2   = 4096   LN out bf16 (dead after in_proj)
#define OFF_XDF   38912    // 32*48*4   = 6144   x_dbl fwd f32 (aliases HNBF)
#define OFF_XDB   45056    // 32*48*4   = 6144   x_dbl bwd f32
#define OFF_COMB  38912    // 32*128*2  = 8192   gated y bf16 (aliases XD after scan)
#define OFF_YC    51200    // 24*128*4  = 12288  combined scan output f32
#define OFF_MO    51200    // 32*64*4   = 8192   out_proj result (aliases YC after gate)
#define SMEM_BYTES 63488

// ---- fast transcendentals: native v_exp_f32 / v_rcp_f32 / v_log_f32 paths ----
__device__ inline float fexp_f(float v)  { return __expf(v); }                 // v_mul + v_exp_f32
__device__ inline float sigm_f(float v)  { return __builtin_amdgcn_rcpf(1.f + __expf(-v)); }
__device__ inline float silu_f(float v)  { return v * sigm_f(v); }
__device__ inline float softplus_f(float v) {
  return (v > 20.f) ? v : __logf(1.f + __expf(v));                             // v_log_f32 path
}

// ---- WMMA fragment helpers (contiguous-K A/B order, ISA C/D layout, wave32) ----
// A 16x32 bf16: lane<16 -> row=lane,   K = k0+0..15 (elems0-15)
//               lane>=16 -> row=lane-16, K = k0+16..31
__device__ inline v16bf load_a_bf16(const bf16_t* A, int stride, int m0, int k0, int lane) {
  int row = m0 + (lane & 15);
  int kb  = k0 + 16 * (lane >> 4);
  const bf16_t* p = A + row * stride + kb;
  v16bf a;
#pragma unroll
  for (int e = 0; e < 16; ++e) a[e] = p[e];
  return a;
}

// B 32x16 bf16, weight stored N-major (n rows, K contiguous):
// lane<16 -> col=n0+lane, K=k0..k0+15 ; lane>=16 -> col=n0+lane-16, K=k0+16..k0+31
__device__ inline v16bf load_b_w(const bf16_t* W, int sk, int n0, int k0, int lane) {
  int n  = n0 + (lane & 15);
  int kb = k0 + 16 * (lane >> 4);
  const bf16_t* p = W + n * sk + kb;
  v16bf b;
#pragma unroll
  for (int e = 0; e < 16; ++e) b[e] = p[e];
  return b;
}

// C/D f32 16x16: element i -> row = m0 + i + 8*(lane>=16), col = n0 + lane%16.
// Destinations are 32-row padded: stores are unconditional (base + imm offsets).
__device__ inline void store_d_f32(float* D, int sd, int m0, int n0, int lane, v8f c) {
  float* p = D + (m0 + 8 * (lane >> 4)) * sd + n0 + (lane & 15);
#pragma unroll
  for (int i = 0; i < 8; ++i) p[i * sd] = c[i];
}

__device__ inline void store_d_bf16(bf16_t* D, int sd, int m0, int n0, int lane, v8f c) {
  bf16_t* p = D + (m0 + 8 * (lane >> 4)) * sd + n0 + (lane & 15);
#pragma unroll
  for (int i = 0; i < 8; ++i) p[i * sd] = (bf16_t)c[i];
}

__device__ inline void store_d_silu_bf16(bf16_t* D, int sd, int m0, int n0, int lane, v8f c) {
  bf16_t* p = D + (m0 + 8 * (lane >> 4)) * sd + n0 + (lane & 15);
#pragma unroll
  for (int i = 0; i < 8; ++i) p[i * sd] = (bf16_t)silu_f(c[i]);
}

// ---- weight prep: f32 -> bf16 (padded), A = -exp(A_log) ----
__global__ void prep_kernel(const float* __restrict__ w_in, const float* __restrict__ w_xpf,
                            const float* __restrict__ w_xpb, const float* __restrict__ w_out,
                            const float* __restrict__ alogf, const float* __restrict__ alogb,
                            bf16_t* __restrict__ Win, bf16_t* __restrict__ Wxpf,
                            bf16_t* __restrict__ Wxpb, bf16_t* __restrict__ Wout,
                            float* __restrict__ Af, float* __restrict__ Ab) {
  int gid = blockIdx.x * blockDim.x + threadIdx.x;
  int gs  = gridDim.x * blockDim.x;
  for (int i = gid; i < 256 * 64; i += gs) Win[i] = (bf16_t)w_in[i];
  for (int i = gid; i < 48 * 128; i += gs) Wxpf[i] = (i < 36 * 128) ? (bf16_t)w_xpf[i] : (bf16_t)0.f;
  for (int i = gid; i < 48 * 128; i += gs) Wxpb[i] = (i < 36 * 128) ? (bf16_t)w_xpb[i] : (bf16_t)0.f;
  for (int i = gid; i < 64 * 128; i += gs) Wout[i] = (bf16_t)w_out[i];
  for (int i = gid; i < 128 * 16; i += gs) { Af[i] = -expf(alogf[i]); Ab[i] = -expf(alogb[i]); }
}

__global__ __launch_bounds__(256) void mamba_kernel(
    const float* __restrict__ x, const float* __restrict__ qk,
    const float* __restrict__ convwf, const float* __restrict__ convbf,
    const float* __restrict__ convwb, const float* __restrict__ convbb,
    const float* __restrict__ dtwf, const float* __restrict__ dtbf,
    const float* __restrict__ dtwb, const float* __restrict__ dtbb,
    const float* __restrict__ Df, const float* __restrict__ Db,
    const float* __restrict__ lng, const float* __restrict__ lnb,
    const bf16_t* __restrict__ Win, const bf16_t* __restrict__ Wxpf,
    const bf16_t* __restrict__ Wxpb, const bf16_t* __restrict__ Wout,
    const float* __restrict__ Afw, const float* __restrict__ Abw,
    float* __restrict__ out) {
  __shared__ __align__(16) unsigned char smem[SMEM_BYTES];
  float*  hb     = (float*)(smem + OFF_H);
  bf16_t* xbf    = (bf16_t*)(smem + OFF_XBF);
  bf16_t* zsil   = (bf16_t*)(smem + OFF_ZSIL);
  bf16_t* uff    = (bf16_t*)(smem + OFF_UF);
  bf16_t* ubb    = (bf16_t*)(smem + OFF_UB);
  bf16_t* hnbf   = (bf16_t*)(smem + OFF_HNBF);
  float*  xdf    = (float*)(smem + OFF_XDF);
  float*  xdb    = (float*)(smem + OFF_XDB);
  bf16_t* combbf = (bf16_t*)(smem + OFF_COMB);
  float*  yc     = (float*)(smem + OFF_YC);
  float*  mo     = (float*)(smem + OFF_MO);

  const int tid  = threadIdx.x;
  const int wv   = tid >> 5;
  const int lane = tid & 31;
  const int seq  = blockIdx.x;
  const int bi   = seq / Nn;
  const int ni   = seq % Nn;

  // h0 = x + qk, reshaped (b,f,n,t) -> (l,f); zero pad rows 24..31 of u buffers once
  for (int i = tid; i < 8 * DIN; i += 256) {
    uff[Tn * DIN + i] = (bf16_t)0.f;
    ubb[Tn * DIN + i] = (bf16_t)0.f;
  }
  for (int i = tid; i < Tn * Fn; i += 256) {
    int l = i >> 6, f = i & 63;
    int g = ((bi * Fn + f) * Nn + ni) * Tn + l;
    hb[i] = x[g] + qk[g];
  }
  __syncthreads();

  for (int blk = 0; blk < 2; ++blk) {
    const float* g0 = lng + (blk * 2 + 0) * Fn;
    const float* b0 = lnb + (blk * 2 + 0) * Fn;
    const float* g1 = lng + (blk * 2 + 1) * Fn;
    const float* b1 = lnb + (blk * 2 + 1) * Fn;

    // P1: LayerNorm(hb) -> hnbf (bf16, rows 24..31 zero-padded)
    if (tid < Tn) {
      const float* r = hb + tid * Fn;
      float m = 0.f;
#pragma unroll 1
      for (int f = 0; f < Fn; ++f) m += r[f];
      m *= (1.f / Fn);
      float v = 0.f;
#pragma unroll 1
      for (int f = 0; f < Fn; ++f) { float d = r[f] - m; v += d * d; }
      v *= (1.f / Fn);
      float rs = rsqrtf(v + 1e-5f);
#pragma unroll 1
      for (int f = 0; f < Fn; ++f)
        hnbf[tid * Fn + f] = (bf16_t)((r[f] - m) * rs * g0[f] + b0[f]);
    } else if (tid < 32) {
#pragma unroll 1
      for (int f = 0; f < Fn; ++f) hnbf[tid * Fn + f] = (bf16_t)0.f;
    }
    __syncthreads();

    // P2: in_proj  (32x256 = A(32x64) x W(64x256)), 32 tiles, 4/wave
#pragma unroll 1
    for (int j = 0; j < 4; ++j) {
      int tt = wv * 4 + j;
      int mt = tt >> 4, nt = tt & 15;
      v8f c = {};
#pragma unroll
      for (int ks = 0; ks < 2; ++ks) {
        v16bf a  = load_a_bf16(hnbf, Fn, mt * 16, ks * 32, lane);
        v16bf bw = load_b_w(Win, Fn, nt * 16, ks * 32, lane);
        c = __builtin_amdgcn_wmma_f32_16x16x32_bf16(false, a, false, bw, (short)0, c, false, false);
      }
      if (nt < 8) store_d_bf16(xbf, DIN, mt * 16, nt * 16, lane, c);             // x
      else        store_d_silu_bf16(zsil, DIN, mt * 16, (nt - 8) * 16, lane, c); // silu(z)
    }
    __syncthreads();

    // P3: depthwise causal conv (fwd + time-flipped bwd) + SiLU -> bf16 u buffers
#pragma unroll 1
    for (int it = 0; it < 12; ++it) {
      int i = tid + it * 256;
      int l = i >> 7, c = i & 127;
      float sf = convbf[c], sb = convbb[c];
#pragma unroll
      for (int j = 0; j < DCONV; ++j) {
        int ls  = l - 3 + j;
        int lsc = (ls < 0) ? 0 : ls;                 // clamp -> unconditional loads
        float wf = (ls >= 0) ? convwf[c * DCONV + j] : 0.f;
        float wb = (ls >= 0) ? convwb[c * DCONV + j] : 0.f;
        sf += wf * (float)xbf[lsc * DIN + c];
        sb += wb * (float)xbf[(23 - lsc) * DIN + c];
      }
      uff[l * DIN + c] = (bf16_t)silu_f(sf);
      ubb[l * DIN + c] = (bf16_t)silu_f(sb);
    }
    __syncthreads();

    // P4: x_proj both directions (24x48 = u(24x128) x W(128x48)), 12 tiles
#pragma unroll 1
    for (int t = wv; t < 12; t += 8) {
      int dir = t / 6, r = t % 6;
      int mt = r / 3, nt = r % 3;
      const bf16_t* Au = dir ? ubb : uff;
      const bf16_t* W  = dir ? Wxpb : Wxpf;
      float*        Dd = dir ? xdb : xdf;
      v8f c = {};
#pragma unroll
      for (int ks = 0; ks < 4; ++ks) {
        v16bf a  = load_a_bf16(Au, DIN, mt * 16, ks * 32, lane);
        v16bf bw = load_b_w(W, DIN, nt * 16, ks * 32, lane);
        c = __builtin_amdgcn_wmma_f32_16x16x32_bf16(false, a, false, bw, (short)0, c, false, false);
      }
      store_d_f32(Dd, 48, mt * 16, nt * 16, lane, c);
    }
    __syncthreads();

    for (int i = tid; i < Tn * DIN; i += 256) yc[i] = 0.f;
    __syncthreads();

    // P5/P6: selective scan, thread = (direction, channel d), 16 states in regs
    {
      int dir = tid >> 7;
      int d   = tid & 127;
      const float*  Aw   = (dir ? Abw : Afw) + d * DST;
      const bf16_t* uloc = dir ? ubb : uff;
      const float*  xd   = dir ? xdb : xdf;
      const float*  dtw  = (dir ? dtwb : dtwf) + d * DTR;
      float dtb = (dir ? dtbb : dtbf)[d];
      float Dv  = (dir ? Db : Df)[d];
      float Ar[DST], h[DST];
#pragma unroll
      for (int s = 0; s < DST; ++s) { Ar[s] = Aw[s]; h[s] = 0.f; }
      float w0 = dtw[0], w1 = dtw[1], w2 = dtw[2], w3 = dtw[3];
#pragma unroll 1
      for (int l = 0; l < Tn; ++l) {
        const float* xr = xd + l * 48;
        float draw  = dtb + w0 * xr[0] + w1 * xr[1] + w2 * xr[2] + w3 * xr[3];
        float delta = softplus_f(draw);
        float ul    = (float)uloc[l * DIN + d];
        float du    = delta * ul;
        float y     = ul * Dv;
#pragma unroll
        for (int s = 0; s < DST; ++s) {
          h[s] = fexp_f(delta * Ar[s]) * h[s] + du * xr[4 + s];   // v_exp_f32 (TRANS)
          y += h[s] * xr[20 + s];
        }
        int lo = dir ? (23 - l) : l;
        atomicAdd(&yc[lo * DIN + d], y);   // fwd + flipped-bwd combine
      }
    }
    __syncthreads();

    // P7: gate by silu(z), pack bf16 A-operand for out_proj (rows 24..31 zero)
#pragma unroll 1
    for (int i = tid; i < 32 * DIN; i += 256) {
      int l = i >> 7, d = i & 127;
      int lc = (l < Tn) ? l : (Tn - 1);          // clamp -> unconditional loads
      float v = yc[lc * DIN + d] * (float)zsil[lc * DIN + d];
      v = (l < Tn) ? v : 0.f;                    // value select, no divergence
      combbf[i] = (bf16_t)v;
    }
    __syncthreads();

    // P8: out_proj (24x64 = comb(24x128) x W(128x64)), 8 tiles, 1/wave
    {
      int mt = wv >> 2, nt = wv & 3;
      v8f c = {};
#pragma unroll
      for (int ks = 0; ks < 4; ++ks) {
        v16bf a  = load_a_bf16(combbf, DIN, mt * 16, ks * 32, lane);
        v16bf bw = load_b_w(Wout, DIN, nt * 16, ks * 32, lane);
        c = __builtin_amdgcn_wmma_f32_16x16x32_bf16(false, a, false, bw, (short)0, c, false, false);
      }
      store_d_f32(mo, Fn, mt * 16, nt * 16, lane, c);
    }
    __syncthreads();

    // P9: hb += LayerNorm(mo)
    if (tid < Tn) {
      const float* r = mo + tid * Fn;
      float m = 0.f;
#pragma unroll 1
      for (int f = 0; f < Fn; ++f) m += r[f];
      m *= (1.f / Fn);
      float v = 0.f;
#pragma unroll 1
      for (int f = 0; f < Fn; ++f) { float d = r[f] - m; v += d * d; }
      v *= (1.f / Fn);
      float rs = rsqrtf(v + 1e-5f);
#pragma unroll 1
      for (int f = 0; f < Fn; ++f)
        hb[tid * Fn + f] += (r[f] - m) * rs * g1[f] + b1[f];
    }
    __syncthreads();
  }

  // final LayerNorm + store (b,f,n*t)
  {
    const float* g4 = lng + 4 * Fn;
    const float* b4 = lnb + 4 * Fn;
    if (tid < Tn) {
      const float* r = hb + tid * Fn;
      float m = 0.f;
#pragma unroll 1
      for (int f = 0; f < Fn; ++f) m += r[f];
      m *= (1.f / Fn);
      float v = 0.f;
#pragma unroll 1
      for (int f = 0; f < Fn; ++f) { float d = r[f] - m; v += d * d; }
      v *= (1.f / Fn);
      float rs = rsqrtf(v + 1e-5f);
#pragma unroll 1
      for (int f = 0; f < Fn; ++f) {
        int g = (bi * Fn + f) * (Nn * Tn) + ni * Tn + tid;
        out[g] = (r[f] - m) * rs * g4[f] + b4[f];
      }
    }
  }
}

extern "C" void kernel_launch(void* const* d_in, const int* in_sizes, int n_in,
                              void* d_out, int out_size, void* d_ws, size_t ws_size,
                              hipStream_t stream) {
  (void)in_sizes; (void)n_in; (void)out_size; (void)ws_size;
  const float* x      = (const float*)d_in[0];
  const float* qk     = (const float*)d_in[1];
  const float* w_in   = (const float*)d_in[2];
  const float* convwf = (const float*)d_in[3];
  const float* convbf = (const float*)d_in[4];
  const float* convwb = (const float*)d_in[5];
  const float* convbb = (const float*)d_in[6];
  const float* xpwf   = (const float*)d_in[7];
  const float* xpwb   = (const float*)d_in[8];
  const float* dtwf   = (const float*)d_in[9];
  const float* dtbf   = (const float*)d_in[10];
  const float* dtwb   = (const float*)d_in[11];
  const float* dtbb   = (const float*)d_in[12];
  const float* alogf  = (const float*)d_in[13];
  const float* alogb  = (const float*)d_in[14];
  const float* Df     = (const float*)d_in[15];
  const float* Db     = (const float*)d_in[16];
  const float* outw   = (const float*)d_in[17];
  const float* lng    = (const float*)d_in[18];
  const float* lnb    = (const float*)d_in[19];

  char* ws = (char*)d_ws;
  bf16_t* Win  = (bf16_t*)(ws + 0);       // 256x64   -> 32768 B
  bf16_t* Wxpf = (bf16_t*)(ws + 32768);   // 48x128   -> 12288 B
  bf16_t* Wxpb = (bf16_t*)(ws + 45056);   // 48x128   -> 12288 B
  bf16_t* Wout = (bf16_t*)(ws + 57344);   // 64x128   -> 16384 B
  float*  Afp  = (float*)(ws + 73728);    // 128x16   ->  8192 B
  float*  Abp  = (float*)(ws + 81920);    // 128x16   ->  8192 B

  prep_kernel<<<64, 256, 0, stream>>>(w_in, xpwf, xpwb, outw, alogf, alogb,
                                      Win, Wxpf, Wxpb, Wout, Afp, Abp);
  mamba_kernel<<<Bn * Nn, 256, 0, stream>>>(x, qk, convwf, convbf, convwb, convbb,
                                            dtwf, dtbf, dtwb, dtbb, Df, Db, lng, lnb,
                                            Win, Wxpf, Wxpb, Wout, Afp, Abp,
                                            (float*)d_out);
}
